// LSTM_24335284699495
// MI455X (gfx1250) — matrix operand order repaired and checked by reading, not compiled, not run on hardware
//
#include <hip/hip_runtime.h>

// ---------------------------------------------------------------------------
// LSTM LM: V=32000, H=1024, O=32000, B=64, T=512
// Recurrent GEMMs in bf16 WMMA (v_wmma_f32_16x16x32_bf16), f32 cell state.
// h-tile staged to LDS via async-to-LDS copies (ASYNCcnt), double-buffered.
// Step kernel: 32 blocks x 8 waves (gate x col-half) -> 2 waves/SIMD32.
// ---------------------------------------------------------------------------

#define Hd 1024
#define Bd 64
#define Td 512
#define Od 32000

typedef __attribute__((ext_vector_type(16))) __bf16 v16bf;
typedef __attribute__((ext_vector_type(8)))  float  v8f;

#if defined(__has_builtin)
#if __has_builtin(__builtin_amdgcn_global_load_async_to_lds_b128)
#define USE_ASYNC_LDS 1
#endif
#endif

#if defined(USE_ASYNC_LDS)
typedef int vint4 __attribute__((ext_vector_type(4)));
typedef __attribute__((address_space(1))) vint4 gvint4;   // global v4i
typedef __attribute__((address_space(3))) vint4 lvint4;   // LDS v4i
#endif

__device__ __forceinline__ void wait_async0() {
#if defined(USE_ASYNC_LDS)
#if __has_builtin(__builtin_amdgcn_s_wait_asynccnt)
    __builtin_amdgcn_s_wait_asynccnt(0);
#else
    asm volatile("s_wait_asynccnt 0x0" ::: "memory");
#endif
#endif
}

__device__ __forceinline__ unsigned short f32_to_bf16(float f) {
    unsigned int u = __float_as_uint(f);
    unsigned int r = u + 0x7FFFu + ((u >> 16) & 1u);   // round-to-nearest-even
    return (unsigned short)(r >> 16);
}

// ---- one-time: Wh (4 gates, each [1024][1024] f32) -> bf16 [4][1024][1024]
__global__ __launch_bounds__(256)
void cvt_wh_kernel(const float* __restrict__ wi, const float* __restrict__ wf,
                   const float* __restrict__ wo, const float* __restrict__ wg,
                   unsigned short* __restrict__ whb) {
    int idx = blockIdx.x * 256 + threadIdx.x;          // 0 .. 4M-1
    int g   = idx >> 20;
    int off = idx & ((1 << 20) - 1);
    const float* s = (g == 0) ? wi : (g == 1) ? wf : (g == 2) ? wo : wg;
    whb[idx] = f32_to_bf16(s[off]);
}

// ---- zero h0 (bf16) and c0 (f32)
__global__ __launch_bounds__(256)
void init_state_kernel(unsigned short* __restrict__ h0, float* __restrict__ c) {
    int i = blockIdx.x * 256 + threadIdx.x;            // 0 .. 65535
    h0[i] = 0;
    c[i]  = 0.0f;
}

// ---- one LSTM timestep.
// grid = 32 blocks (one per 32-wide hidden column tile), 256 threads = 8 waves.
// wave w: gate = w&3, colHalf = w>>2; each wave does 4 batch row-tiles.
__global__ __launch_bounds__(256)
void lstm_step_kernel(const int* __restrict__ x, int t,
                      const float* __restrict__ Ei, const float* __restrict__ Ef,
                      const float* __restrict__ Eo, const float* __restrict__ Eg,
                      const float* __restrict__ bi, const float* __restrict__ bfv,
                      const float* __restrict__ bo, const float* __restrict__ bg,
                      const unsigned short* __restrict__ Whb,
                      const unsigned short* __restrict__ h_in,
                      unsigned short* __restrict__ h_out,
                      float* __restrict__ c) {
    // Phase 1: double-buffered h slab  h[:, kk..kk+128) as [64][128] bf16 (16KB x2)
    // Phase 2 (after barrier): gate tiles for the fused cell update (32 KB)
    __shared__ union {
        uint4 hTile4[2][1024];             // 32 KB
        float gLds[2][4][4][8][32];        // 32 KB  [colHalf][gate][rt][r][lane]
    } smem;

    const int tid     = threadIdx.x;
    const int wave    = tid >> 5;          // 0..7
    const int gate    = wave & 3;          // i,f,o,g
    const int colHalf = wave >> 2;         // 0/1
    const int lane    = tid & 31;
    const int lsub    = lane & 15;
    const int lhalf   = lane >> 4;         // 0/1
    const int nBase   = blockIdx.x * 32 + colHalf * 16;
    const int nl      = nBase + lsub;      // hidden column 0..1023

    const float* Ew = (gate == 0) ? Ei : (gate == 1) ? Ef : (gate == 2) ? Eo : Eg;
    const float* bw = (gate == 0) ? bi : (gate == 1) ? bfv : (gate == 2) ? bo : bg;

    // slab stager: copies h_in[:, kk..kk+128) into smem.hTile4[buf]
    auto stage = [&](int buf, int kk) {
        for (int u = tid; u < 1024; u += 256) {
            int m  = u >> 4;               // 64 rows x 16 uint4 (128 bf16) per row
            int k8 = u & 15;
            const unsigned short* gp = h_in + m * Hd + kk + k8 * 8;
#if defined(USE_ASYNC_LDS)
            __builtin_amdgcn_global_load_async_to_lds_b128(
                (gvint4*)gp, (lvint4*)&smem.hTile4[buf][u], 0, 0);
#else
            smem.hTile4[buf][u] = *(const uint4*)gp;
#endif
        }
    };

    stage(0, 0);   // prefetch first slab (overlaps with E-gather below)

    // C init = E[tok] + bias, using the 16x16 f32 C/D lane layout.
    v8f acc[4];
    const float bias = bw[nl];
#pragma unroll
    for (int rt = 0; rt < 4; ++rt) {
#pragma unroll
        for (int r = 0; r < 8; ++r) {
            int m   = rt * 16 + lhalf * 8 + r;
            int tok = x[m * Td + t];
            acc[rt][r] = Ew[(long)tok * Hd + nl] + bias;
        }
    }

    // B operand base: Wh row (gate, hidden col), contiguous K; upper half-wave K+16.
    const unsigned short* wrow =
        Whb + ((long)(gate * Hd + nBase + lsub)) * Hd + lhalf * 16;

#pragma unroll 1
    for (int slab = 0; slab < 8; ++slab) {
        const int kk  = slab * 128;
        const int buf = slab & 1;
        wait_async0();                       // my slab copies landed in LDS
        __syncthreads();                     // everyone's landed / prior reads done
        if (slab < 7) {
            stage(buf ^ 1, kk + 128);        // async-prefetch next slab
            __builtin_prefetch(wrow + kk + 128, 0, 0);   // global_prefetch_b8
        }
        const unsigned short* hT = (const unsigned short*)smem.hTile4[buf];
#pragma unroll
        for (int k0 = 0; k0 < 128; k0 += 32) {
            // B: 32 contiguous bf16 of this Wh row (per-lane 32B load)
            union { v16bf v; uint4 u[2]; } b;
            const uint4* bp = (const uint4*)(wrow + kk + k0);
            b.u[0] = bp[0];
            b.u[1] = bp[1];
#pragma unroll
            for (int rt = 0; rt < 4; ++rt) {
                // A: lanes 0-15 -> K {0..15}; lanes 16-31 -> K {16..31}
                union { v16bf v; uint4 u[2]; } a;
                const unsigned short* ap =
                    hT + (rt * 16 + lsub) * 128 + k0 + lhalf * 16;
                a.u[0] = *(const uint4*)ap;          // ds_load_b128
                a.u[1] = *(const uint4*)(ap + 8);
                acc[rt] = __builtin_amdgcn_wmma_f32_16x16x32_bf16(
                    false, a.v, false, b.v, (short)0, acc[rt], false, false);
            }
        }
    }

    // hTile no longer needed -> reuse the same LDS for the gate exchange
    __syncthreads();
#pragma unroll
    for (int rt = 0; rt < 4; ++rt)
#pragma unroll
        for (int r = 0; r < 8; ++r)
            smem.gLds[colHalf][gate][rt][r][lane] = acc[rt][r];
    __syncthreads();

    // fused cell update: 64 rows x 32 cols = 2048 elements, 8 per thread
#pragma unroll
    for (int j = 0; j < 8; ++j) {
        int e    = tid + j * 256;        // 0..2047
        int m    = e >> 5;               // batch row 0..63
        int col5 = e & 31;               // column within block tile
        int ch   = col5 >> 4;
        int nn   = col5 & 15;
        int rt = m >> 4, mr = m & 15, r = mr & 7, lh = mr >> 3;
        int l  = lh * 16 + nn;
        float iv = smem.gLds[ch][0][rt][r][l];
        float fv = smem.gLds[ch][1][rt][r][l];
        float ov = smem.gLds[ch][2][rt][r][l];
        float gv = smem.gLds[ch][3][rt][r][l];
        float si = 1.0f / (1.0f + __expf(-iv));
        float sf = 1.0f / (1.0f + __expf(-fv));
        float so = 1.0f / (1.0f + __expf(-ov));
        float tg = tanhf(gv);
        long  ix = (long)m * Hd + blockIdx.x * 32 + col5;
        float cn = sf * c[ix] + si * tg;
        float hn = so * tanhf(cn);
        c[ix]    = cn;
        h_out[ix] = f32_to_bf16(hn);
    }
}

// ---- logits: out[64][32000] = h @ Why^T + by ; Why converted f32->bf16 on the fly
// grid = (500, 4): blockIdx.y = batch row tile, each of 4 waves owns one 16-col tile.
__global__ __launch_bounds__(128)
void logits_kernel(const unsigned short* __restrict__ hfin,
                   const float* __restrict__ Why, const float* __restrict__ by,
                   float* __restrict__ out) {
    const int wave  = threadIdx.x >> 5;
    const int lane  = threadIdx.x & 31;
    const int lsub  = lane & 15;
    const int lhalf = lane >> 4;
    const int rowBase = blockIdx.y * 16;
    const int n = (blockIdx.x * 4 + wave) * 16 + lsub;

    v8f acc;
    const float bias = by[n];
#pragma unroll
    for (int r = 0; r < 8; ++r) acc[r] = bias;

    const unsigned short* arow = hfin + (rowBase + lsub) * Hd + lhalf * 16;
    const float* wr = Why + (long)n * Hd + lhalf * 16;

    for (int k0 = 0; k0 < Hd; k0 += 32) {
        union { v16bf v; uint4 u[2]; } a;
        a.u[0] = *(const uint4*)(arow + k0);
        a.u[1] = *(const uint4*)(arow + k0 + 8);
        union { v16bf v; unsigned short s[16]; } b;
        const float* w = wr + k0;
#pragma unroll
        for (int i = 0; i < 16; ++i) b.s[i] = f32_to_bf16(w[i]);
        acc = __builtin_amdgcn_wmma_f32_16x16x32_bf16(
            false, a.v, false, b.v, (short)0, acc, false, false);
    }
#pragma unroll
    for (int r = 0; r < 8; ++r) {
        int m = rowBase + lhalf * 8 + r;
        out[(long)m * Od + n] = acc[r];
    }
}

// ---------------------------------------------------------------------------
extern "C" void kernel_launch(void* const* d_in, const int* in_sizes, int n_in,
                              void* d_out, int out_size, void* d_ws, size_t ws_size,
                              hipStream_t stream) {
    const int*   x   = (const int*)d_in[0];
    const float* Ei  = (const float*)d_in[1];
    const float* Ef  = (const float*)d_in[2];
    const float* Eo  = (const float*)d_in[3];
    const float* Eg  = (const float*)d_in[4];
    const float* Wi  = (const float*)d_in[5];
    const float* bi  = (const float*)d_in[6];
    const float* Wf  = (const float*)d_in[7];
    const float* bfv = (const float*)d_in[8];
    const float* Wo  = (const float*)d_in[9];
    const float* bo  = (const float*)d_in[10];
    const float* Wg  = (const float*)d_in[11];
    const float* bg  = (const float*)d_in[12];
    const float* Why = (const float*)d_in[13];
    const float* by  = (const float*)d_in[14];

    // workspace layout (needs ~8.9 MB)
    char* ws = (char*)d_ws;
    unsigned short* Whb   = (unsigned short*)ws;                       // 8 MB bf16 Wh
    unsigned short* hbuf0 = (unsigned short*)(ws + (size_t)8 * 1024 * 1024);
    unsigned short* hbuf1 = hbuf0 + Bd * Hd;                           // 128 KB each
    float*          c     = (float*)(ws + (size_t)8 * 1024 * 1024
                                        + (size_t)2 * Bd * Hd * 2);    // 256 KB

    cvt_wh_kernel<<<dim3(16384), dim3(256), 0, stream>>>(Wi, Wf, Wo, Wg, Whb);
    init_state_kernel<<<dim3(256), dim3(256), 0, stream>>>(hbuf0, c);

    unsigned short* hb[2] = {hbuf0, hbuf1};
    for (int t = 0; t < Td; ++t) {
        lstm_step_kernel<<<dim3(32), dim3(256), 0, stream>>>(
            x, t, Ei, Ef, Eo, Eg, bi, bfv, bo, bg,
            Whb, hb[t & 1], hb[(t + 1) & 1], c);
    }
    // T even -> final h is in hbuf0
    logits_kernel<<<dim3(500, 4), dim3(128), 0, stream>>>(hbuf0, Why, by, (float*)d_out);
}
